// SupConLoss_65824668778816
// MI455X (gfx1250) — matrix operand order repaired and checked by reading, not compiled, not run on hardware
//
#include <hip/hip_runtime.h>
#include <hip/hip_bf16.h>
#include <math.h>

// Problem constants (from setup_inputs)
#define NB   8000          // reps rows
#define NC   8             // classes
#define NK   24            // anchors per class
#define ND   256           // feature dim
#define NN   8192          // NB + NC*NK
#define EPS  1e-8f

typedef __attribute__((ext_vector_type(16))) _Float16 v16h;
typedef __attribute__((ext_vector_type(8)))  _Float16 v8h;
typedef __attribute__((ext_vector_type(8)))  float    v8f;

union V16u { v16h v; struct { v8h lo, hi; } p; };

// ---------------------------------------------------------------- zero scratch
__global__ void __launch_bounds__(256) zeroK(float* posRow, float* negRow,
                                             int* classCnt, float* scal) {
  int i = blockIdx.x * blockDim.x + threadIdx.x;
  if (i < NN) { posRow[i] = 0.f; negRow[i] = 0.f; }
  if (i < NC) classCnt[i] = 0;
  if (i < 2)  scal[i] = 0.f;
}

// ------------------------------------------------- normalize rows -> f16 panel
// one wave per row; 8 waves (256 threads) per block
__global__ void __launch_bounds__(256) prepK(const float* __restrict__ reps,
                                             const float* __restrict__ emo,
                                             const int*   __restrict__ labels,
                                             const int*   __restrict__ albl,
                                             _Float16* __restrict__ rnh,
                                             int* __restrict__ cl,
                                             int* __restrict__ classCnt) {
  const int lane = threadIdx.x & 31;
  const int wave = threadIdx.x >> 5;
  const int row  = blockIdx.x * 8 + wave;
  const float* src = (row < NB) ? (reps + (size_t)row * ND)
                                : (emo + (size_t)(row - NB) * ND);
  float x[8];
  float ss = 0.f;
#pragma unroll
  for (int e = 0; e < 8; ++e) { x[e] = src[lane + e * 32]; ss += x[e] * x[e]; }
#pragma unroll
  for (int off = 16; off > 0; off >>= 1) ss += __shfl_xor(ss, off, 32);
  const float inv = 1.f / fmaxf(sqrtf(ss), 1e-8f);
  _Float16* dst = rnh + (size_t)row * ND;
#pragma unroll
  for (int e = 0; e < 8; ++e) dst[lane + e * 32] = (_Float16)(x[e] * inv);
  if (lane == 0) {
    const int lbl = (row < NB) ? labels[row] : albl[row - NB];
    cl[row] = lbl;
    atomicAdd(&classCnt[lbl], 1);
  }
}

// -------------------------------------------------------------- Gram + epilogue
// grid 512: blockRow = bx>>3 (128-row panel), jslice = bx&7 (64 j-tiles each)
// wave w owns rows rowBase..rowBase+15; A panel (16x256 f16) kept in 64 VGPRs.
// B stream is software-pipelined at half-tile (4 k-step) granularity so each
// WMMA group waits only on loads issued a full group earlier.
__global__ void __launch_bounds__(256) gramK(const _Float16* __restrict__ rnh,
                                             const int* __restrict__ cl,
                                             float* __restrict__ posRow,
                                             float* __restrict__ negRow) {
  const int lane = threadIdx.x & 31;
  const int wave = threadIdx.x >> 5;
  const int blockRow = blockIdx.x >> 3;
  const int jslice   = blockIdx.x & 7;
  const int rowBase  = blockRow * 128 + wave * 16;
  const int hi = lane >> 4;        // lane group (ISA fragment layout)
  const int nl = lane & 15;
  const int jtile0 = jslice * 64;

  // A fragments: contiguous k order used here, same as B (lanes 0-15: K 0..15; 16-31: K 16..31)
  v16h a[8];
  {
    const _Float16* rp = rnh + (size_t)(rowBase + nl) * ND;
#pragma unroll
    for (int s = 0; s < 8; ++s) {
      const int k0 = s * 32;
      V16u u;
      u.v = *(const v16h*)(rp + k0 + hi * 16);
      a[s] = u.v;
    }
  }
  int cli[8];
#pragma unroll
  for (int r = 0; r < 8; ++r) cli[r] = cl[rowBase + r + hi * 8];

  float pos[8], neg[8];
#pragma unroll
  for (int r = 0; r < 8; ++r) { pos[r] = 0.f; neg[r] = 0.f; }

  // B fragment: 32x16, lane holds N=nl, 16 consecutive K (hi selects K half);
  // half h covers k-steps 4h..4h+3 (128 halves).
  auto loadHalf = [&](int t, int h, v16h (&b)[4]) {
    const int jbase = (jtile0 + t) * 16;
    const _Float16* bp = rnh + (size_t)(jbase + nl) * ND + hi * 16 + h * 128;
#pragma unroll
    for (int s = 0; s < 4; ++s) b[s] = *(const v16h*)(bp + s * 32);
  };

  v16h bx[4], by[4];
  int cljx, cljy = 0;
  loadHalf(0, 0, bx);
  cljx = cl[jtile0 * 16 + nl];

  for (int t = 0; t < 64; ++t) {
    loadHalf(t, 1, by);                                  // 2nd half of tile t
    if (t != 63) cljy = cl[(jtile0 + t + 1) * 16 + nl];  // labels, tile t+1
    v8f c = {};
#pragma unroll
    for (int s = 0; s < 4; ++s)
      c = __builtin_amdgcn_wmma_f32_16x16x32_f16(false, a[s], false, bx[s],
                                                 (short)0, c, false, false);
    if (t != 63) loadHalf(t + 1, 0, bx);                 // 1st half of tile t+1
#pragma unroll
    for (int s = 0; s < 4; ++s)
      c = __builtin_amdgcn_wmma_f32_16x16x32_f16(false, a[4 + s], false, by[s],
                                                 (short)0, c, false, false);
    const int jcol = (jtile0 + t) * 16 + nl;
#pragma unroll
    for (int r = 0; r < 8; ++r) {
      const int irow = rowBase + r + hi * 8;             // C layout: M=r(+8), N=nl
      const float e = __expf(5.f * c[r] + 5.0000001f);   // (1+cos)/2/T + eps/T
      const bool same = (cli[r] == cljx);
      pos[r] += (same && (irow != jcol)) ? e : 0.f;
      neg[r] += same ? 0.f : e;
    }
    cljx = cljy;
  }
  // reduce across the 16 lanes sharing each M (xor masks stay in-group)
#pragma unroll
  for (int r = 0; r < 8; ++r) {
    float p = pos[r], n = neg[r];
#pragma unroll
    for (int off = 8; off > 0; off >>= 1) {
      p += __shfl_xor(p, off, 32);
      n += __shfl_xor(n, off, 32);
    }
    pos[r] = p; neg[r] = n;
  }
  if (nl == 0) {
#pragma unroll
    for (int r = 0; r < 8; ++r) {
      atomicAdd(&posRow[rowBase + r + hi * 8], pos[r]);
      atomicAdd(&negRow[rowBase + r + hi * 8], neg[r]);
    }
  }
}

// ------------------------------------------------------------- per-row loss
__global__ void __launch_bounds__(256) rowlossK(const float* __restrict__ posRow,
                                                const float* __restrict__ negRow,
                                                const int* __restrict__ cl,
                                                const int* __restrict__ classCnt,
                                                float* __restrict__ scal) {
  const int i = blockIdx.x * blockDim.x + threadIdx.x;
  if (i >= NN) return;
  const float pos = posRow[i], neg = negRow[i];
  const float cnt = (float)(classCnt[cl[i]] - 1);
  float probs = pos / (pos + neg);
  probs = probs / (cnt + EPS);
  const float lv = -logf(probs + EPS);
  if (lv > 0.f) {
    atomicAdd(&scal[0], lv);
    atomicAdd(&scal[1], 1.f);
  }
}

// ------------------------------------------- angle loss (8x8) + final combine
__global__ void __launch_bounds__(256) finalizeK(const float* __restrict__ emo,
                                                 const float* __restrict__ scal,
                                                 float* __restrict__ out) {
  __shared__ float cm[NC * ND];
  __shared__ float meanD[ND];
  __shared__ float invn[NC];
  __shared__ float cosm[NC * NC];
  __shared__ float racos[NC];
  const int t = threadIdx.x;
  // class means over K
#pragma unroll
  for (int q = 0; q < 8; ++q) {
    const int v = t + q * 256, c = v >> 8, d = v & 255;
    float s = 0.f;
    for (int k = 0; k < NK; ++k) s += emo[((size_t)(c * NK + k)) * ND + d];
    cm[v] = s * (1.f / (float)NK);
  }
  __syncthreads();
  { float s = 0.f;
    for (int c = 0; c < NC; ++c) s += cm[c * ND + t];
    meanD[t] = s * (1.f / (float)NC); }
  __syncthreads();
#pragma unroll
  for (int q = 0; q < 8; ++q) { const int v = t + q * 256; cm[v] -= meanD[v & 255]; }
  __syncthreads();
  if (t < NC) {
    float s = 0.f;
    for (int d = 0; d < ND; ++d) { const float x = cm[t * ND + d]; s += x * x; }
    invn[t] = 1.f / fmaxf(sqrtf(s), 1e-12f);
  }
  __syncthreads();
#pragma unroll
  for (int q = 0; q < 8; ++q) { const int v = t + q * 256; cm[v] *= invn[v >> 8]; }
  __syncthreads();
  if (t < NC * NC) {
    const int ai = t >> 3, bi = t & 7;
    float s = 0.f;
    for (int d = 0; d < ND; ++d) s += cm[ai * ND + d] * cm[bi * ND + d];
    cosm[t] = (ai == bi) ? -s : s;     // cosine - 2*diag(diag(cosine))
  }
  __syncthreads();
  if (t < NC) {
    float mx = -2.f;
    for (int b = 0; b < NC; ++b) mx = fmaxf(mx, cosm[t * NC + b]);
    racos[t] = acosf(fminf(fmaxf(mx, -0.99999f), 0.99999f));
  }
  __syncthreads();
  if (t == 0) {
    float gmx = -2.f;
    for (int i = 0; i < NC * NC; ++i) gmx = fmaxf(gmx, cosm[i]);
    const float max_cosine = fminf(fmaxf(gmx, -0.99999f), 0.99999f);
    float s = 0.f;
    for (int r = 0; r < NC; ++r) s += racos[r];
    const float angle = -(s * (1.f / (float)NC));
    out[0] = scal[0] / (scal[1] + EPS) + angle;
    out[1] = max_cosine;
  }
}

extern "C" void kernel_launch(void* const* d_in, const int* in_sizes, int n_in,
                              void* d_out, int out_size, void* d_ws, size_t ws_size,
                              hipStream_t stream) {
  const float* reps   = (const float*)d_in[0];   // [8000,256] f32
  const int*   labels = (const int*)  d_in[1];   // [8000]
  const float* emo    = (const float*)d_in[2];   // [8,24,256] f32
  const int*   albl   = (const int*)  d_in[3];   // [192]

  char* ws = (char*)d_ws;
  _Float16* rnh    = (_Float16*)ws;                               // 4 MB
  int*      cl     = (int*)  (ws + (size_t)NN * ND * 2);          // 32 KB
  float*    posRow = (float*)(ws + (size_t)NN * ND * 2 + 32768);  // 32 KB
  float*    negRow = (float*)(ws + (size_t)NN * ND * 2 + 65536);  // 32 KB
  int*   classCnt  = (int*)  (ws + (size_t)NN * ND * 2 + 98304);
  float* scal      = (float*)(ws + (size_t)NN * ND * 2 + 98304 + 64);

  zeroK    <<<NN / 256, 256, 0, stream>>>(posRow, negRow, classCnt, scal);
  prepK    <<<NN / 8,   256, 0, stream>>>(reps, emo, labels, albl, rnh, cl, classCnt);
  gramK    <<<512,      256, 0, stream>>>(rnh, cl, posRow, negRow);
  rowlossK <<<NN / 256, 256, 0, stream>>>(posRow, negRow, cl, classCnt, scal);
  finalizeK<<<1,        256, 0, stream>>>(emo, scal, (float*)d_out);
}
